// StandaloneTDATR_50251117363621
// MI455X (gfx1250) — matrix operand order repaired and checked by reading, not compiled, not run on hardware
//
#include <hip/hip_runtime.h>
#include <hip/hip_bf16.h>
#include <math.h>

// ---------------------------------------------------------------------------
// Types / helpers
// ---------------------------------------------------------------------------
typedef __attribute__((ext_vector_type(16))) __bf16 v16bf;
typedef __attribute__((ext_vector_type(8)))  float  v8f;

union FragU {
  v16bf v;
  uint4 q[2];
  unsigned short u[16];
};

__device__ __forceinline__ unsigned short f2bf(float f) {
  unsigned int u = __float_as_uint(f);
  u += 0x7FFFu + ((u >> 16) & 1u);              // round-to-nearest-even
  return (unsigned short)(u >> 16);
}

__device__ __forceinline__ v8f wmma_bf16(v16bf a, v16bf b, v8f c) {
  return __builtin_amdgcn_wmma_f32_16x16x32_bf16(false, a, false, b,
                                                 (short)0, c, false, false);
}

// A fragment (16x32, bf16) from row-major [M,K]:
// lane m = l&15; K block = 16*(l>>4), 16 contiguous K values
__device__ __forceinline__ v16bf frag_A(const unsigned short* base, int ld,
                                        int k0, int lane) {
  const int r  = lane & 15;
  const int kb = k0 + 16 * (lane >> 4);
  const unsigned short* p = base + (size_t)r * ld + kb;
  FragU f;
  f.q[0] = *(const uint4*)(p);
  f.q[1] = *(const uint4*)(p + 8);
  return f.v;
}

// B fragment (32x16, bf16) from row-major [N,K] (i.e. computing A @ W^T):
// lane n = l&15; K block = 16*(l>>4), 16 contiguous K values
__device__ __forceinline__ v16bf frag_B_NK(const unsigned short* base, int ld,
                                           int k0, int lane) {
  const int n  = lane & 15;
  const int kb = k0 + 16 * (lane >> 4);
  const unsigned short* p = base + (size_t)n * ld + kb;
  FragU f;
  f.q[0] = *(const uint4*)(p);
  f.q[1] = *(const uint4*)(p + 8);
  return f.v;
}

// B fragment (32x16, bf16) from row-major [K,N] (e.g. V matrix): strided
__device__ __forceinline__ v16bf frag_B_KN(const unsigned short* base, int ld,
                                           int k0, int lane) {
  const int n  = lane & 15;
  const int kb = k0 + 16 * (lane >> 4);
  FragU f;
#pragma unroll
  for (int e = 0; e < 16; ++e) f.u[e] = base[(size_t)(kb + e) * ld + n];
  return f.v;
}

__device__ __forceinline__ float redmax16(float v) {
#pragma unroll
  for (int off = 1; off < 16; off <<= 1) v = fmaxf(v, __shfl_xor(v, off, 32));
  return v;
}
__device__ __forceinline__ float redsum16(float v) {
#pragma unroll
  for (int off = 1; off < 16; off <<= 1) v += __shfl_xor(v, off, 32);
  return v;
}

// ---------------------------------------------------------------------------
// Generic bf16 GEMM: C[M,N] = A[M,K] @ W[N,K]^T (+bias, activation)
// block = 128 threads = 4 waves, each wave 64x64, block tile 128x128
// ACT: 0 none, 1 exact GELU, 2 ReLU, 3 clip(-32,32)
// ---------------------------------------------------------------------------
template <int ACT, int WRITE_F32, int WRITE_BF16>
__global__ __launch_bounds__(128) void gemm_bf16(
    const unsigned short* __restrict__ A, const unsigned short* __restrict__ W,
    const float* __restrict__ bias, float* __restrict__ Cf,
    unsigned short* __restrict__ Cb, int M, int N, int K) {
  const int lane = threadIdx.x & 31;
  const int wave = threadIdx.x >> 5;
  const int m0 = blockIdx.x * 128 + (wave >> 1) * 64;
  const int n0 = blockIdx.y * 128 + (wave & 1) * 64;
  const v8f vz = {0.f, 0.f, 0.f, 0.f, 0.f, 0.f, 0.f, 0.f};
  v8f acc[4][4];
#pragma unroll
  for (int i = 0; i < 4; ++i)
#pragma unroll
    for (int j = 0; j < 4; ++j) acc[i][j] = vz;

  for (int k0 = 0; k0 < K; k0 += 32) {
    if (k0 + 32 < K)
      __builtin_prefetch(A + (size_t)(m0 + (lane & 15)) * K + k0 + 32, 0, 1);
    v16bf a[4], b[4];
#pragma unroll
    for (int i = 0; i < 4; ++i)
      a[i] = frag_A(A + (size_t)(m0 + 16 * i) * K, K, k0, lane);
#pragma unroll
    for (int j = 0; j < 4; ++j)
      b[j] = frag_B_NK(W + (size_t)(n0 + 16 * j) * K, K, k0, lane);
#pragma unroll
    for (int i = 0; i < 4; ++i)
#pragma unroll
      for (int j = 0; j < 4; ++j) acc[i][j] = wmma_bf16(a[i], b[j], acc[i][j]);
  }

  const int cn = lane & 15, hi = lane >> 4;
#pragma unroll
  for (int j = 0; j < 4; ++j) {
    const int col = n0 + 16 * j + cn;
    const float bs = bias ? bias[col] : 0.f;
#pragma unroll
    for (int i = 0; i < 4; ++i) {
#pragma unroll
      for (int r = 0; r < 8; ++r) {
        const int row = m0 + 16 * i + r + 8 * hi;
        float v = acc[i][j][r] + bs;
        if (ACT == 1) v = 0.5f * v * (1.f + erff(v * 0.70710678118654752f));
        else if (ACT == 2) v = fmaxf(v, 0.f);
        else if (ACT == 3) v = fminf(fmaxf(v, -32.f), 32.f);
        if (WRITE_F32)  Cf[(size_t)row * N + col] = v;
        if (WRITE_BF16) Cb[(size_t)row * N + col] = f2bf(v);
      }
    }
  }
}

// ---------------------------------------------------------------------------
// LayerNorm (C=256) -> bf16, optional Swin window-partition row permutation
// ---------------------------------------------------------------------------
__global__ __launch_bounds__(256) void ln_to_bf16(
    const float* __restrict__ x, const float* __restrict__ g,
    const float* __restrict__ b, unsigned short* __restrict__ out,
    int window_mode) {
  __shared__ float red[256];
  const int row = blockIdx.x, c = threadIdx.x;
  const float v = x[(size_t)row * 256 + c];
  red[c] = v;
  __syncthreads();
#pragma unroll
  for (int s = 128; s > 0; s >>= 1) {
    if (c < s) red[c] += red[c + s];
    __syncthreads();
  }
  const float mean = red[0] * (1.f / 256.f);
  __syncthreads();
  const float d = v - mean;
  red[c] = d * d;
  __syncthreads();
#pragma unroll
  for (int s = 128; s > 0; s >>= 1) {
    if (c < s) red[c] += red[c + s];
    __syncthreads();
  }
  const float var = red[0] * (1.f / 256.f);
  const float y = d * rsqrtf(var + 1e-5f) * g[c] + b[c];
  int orow = row;
  if (window_mode) {                       // (b,y,x) -> window-major row
    const int bb = row >> 14, l = row & 16383;
    const int yy = l >> 7, xx = l & 127;
    orow = ((bb << 8) + ((yy >> 3) << 4) + (xx >> 3)) * 64 +
           (yy & 7) * 8 + (xx & 7);
  }
  out[(size_t)orow * 256 + c] = f2bf(y);
}

// ---------------------------------------------------------------------------
// Windowed self-attention per (window, head). QKV[32768,768] bf16.
// S = QK^T (WMMA) + rel-pos bias, fp32 softmax, P staged via LDS, O = P V.
// ---------------------------------------------------------------------------
__global__ __launch_bounds__(32) void win_attn(
    const unsigned short* __restrict__ QKV, const float* __restrict__ rbias,
    unsigned short* __restrict__ O) {
  __shared__ unsigned short plds[64 * 64];
  const int lane = threadIdx.x;
  const int w = blockIdx.x >> 3, h = blockIdx.x & 7;
  const unsigned short* qb = QKV + (size_t)w * 64 * 768 + h * 32;
  const unsigned short* kbp = qb + 256;
  const unsigned short* vb = qb + 512;
  const v8f vz = {0.f, 0.f, 0.f, 0.f, 0.f, 0.f, 0.f, 0.f};

  v16bf aq[4], bk[4];
#pragma unroll
  for (int i = 0; i < 4; ++i)
    aq[i] = frag_A(qb + (size_t)(16 * i) * 768, 768, 0, lane);
#pragma unroll
  for (int j = 0; j < 4; ++j)
    bk[j] = frag_B_NK(kbp + (size_t)(16 * j) * 768, 768, 0, lane);

  v8f s[4][4];
#pragma unroll
  for (int i = 0; i < 4; ++i)
#pragma unroll
    for (int j = 0; j < 4; ++j) s[i][j] = wmma_bf16(aq[i], bk[j], vz);

  const int cn = lane & 15, hi = lane >> 4;
  const float scale = 0.17677669529663689f;  // 1/sqrt(32)
#pragma unroll
  for (int i = 0; i < 4; ++i)
#pragma unroll
    for (int r = 0; r < 8; ++r) {
      const int m = 16 * i + r + 8 * hi;
#pragma unroll
      for (int j = 0; j < 4; ++j) {
        const int kk = 16 * j + cn;
        const int rel = ((m >> 3) - (kk >> 3) + 7) * 15 +
                        ((m & 7) - (kk & 7) + 7);
        s[i][j][r] = s[i][j][r] * scale + rbias[rel * 8 + h];
      }
    }
  // fp32 softmax over 64 keys per row
#pragma unroll
  for (int i = 0; i < 4; ++i)
#pragma unroll
    for (int r = 0; r < 8; ++r) {
      float mx = s[i][0][r];
#pragma unroll
      for (int j = 1; j < 4; ++j) mx = fmaxf(mx, s[i][j][r]);
      mx = redmax16(mx);
      float e[4], sum = 0.f;
#pragma unroll
      for (int j = 0; j < 4; ++j) {
        e[j] = __expf(s[i][j][r] - mx);
        sum += e[j];
      }
      sum = redsum16(sum);
      const float inv = 1.f / sum;
      const int m = 16 * i + r + 8 * hi;
#pragma unroll
      for (int j = 0; j < 4; ++j)
        plds[m * 64 + 16 * j + cn] = f2bf(e[j] * inv);
    }
  __syncthreads();

  v8f o[4][2];
#pragma unroll
  for (int i = 0; i < 4; ++i) { o[i][0] = vz; o[i][1] = vz; }
#pragma unroll
  for (int ks = 0; ks < 2; ++ks) {
    const int k0 = ks * 32;
    v16bf ap[4], bv[2];
#pragma unroll
    for (int i = 0; i < 4; ++i) ap[i] = frag_A(plds + 16 * i * 64, 64, k0, lane);
#pragma unroll
    for (int n = 0; n < 2; ++n) bv[n] = frag_B_KN(vb + 16 * n, 768, k0, lane);
#pragma unroll
    for (int i = 0; i < 4; ++i)
#pragma unroll
      for (int n = 0; n < 2; ++n) o[i][n] = wmma_bf16(ap[i], bv[n], o[i][n]);
  }
#pragma unroll
  for (int i = 0; i < 4; ++i)
#pragma unroll
    for (int n = 0; n < 2; ++n)
#pragma unroll
      for (int r = 0; r < 8; ++r) {
        const int m = 16 * i + r + 8 * hi;
        O[(size_t)(w * 64 + m) * 256 + h * 32 + n * 16 + cn] =
            f2bf(o[i][n][r]);
      }
}

// ---------------------------------------------------------------------------
// Flash-style cross attention: per (b, head, 16-row q tile), stream 16384 keys
// Q[1024,256] rows b*512+t; KV[2*16384,256] bf16 (= final x); CTX bf16 out.
// ---------------------------------------------------------------------------
__global__ __launch_bounds__(32) void cross_attn(
    const unsigned short* __restrict__ Q, const unsigned short* __restrict__ KV,
    unsigned short* __restrict__ CTX) {
  __shared__ unsigned short plds[16 * 64];
  const int lane = threadIdx.x;
  const int qt = blockIdx.x & 31;
  const int h = (blockIdx.x >> 5) & 7;
  const int b = blockIdx.x >> 8;
  const unsigned short* qbase =
      Q + (size_t)(b * 512 + qt * 16) * 256 + h * 32;
  const v16bf aq = frag_A(qbase, 256, 0, lane);
  const int cn = lane & 15, hi = lane >> 4;
  const float scale = 0.17677669529663689f;
  const v8f vz = {0.f, 0.f, 0.f, 0.f, 0.f, 0.f, 0.f, 0.f};
  float mrun[8], lrun[8];
#pragma unroll
  for (int r = 0; r < 8; ++r) { mrun[r] = -1e30f; lrun[r] = 0.f; }
  v8f o[2] = {vz, vz};

  for (int kc = 0; kc < 16384; kc += 64) {
    const unsigned short* kbase =
        KV + (size_t)(b * 16384 + kc) * 256 + h * 32;
    v8f s[4] = {vz, vz, vz, vz};
#pragma unroll
    for (int j = 0; j < 4; ++j) {
      v16bf bk = frag_B_NK(kbase + (size_t)(16 * j) * 256, 256, 0, lane);
      s[j] = wmma_bf16(aq, bk, s[j]);
    }
#pragma unroll
    for (int r = 0; r < 8; ++r) {
      float mx = s[0][r] * scale;
#pragma unroll
      for (int j = 1; j < 4; ++j) mx = fmaxf(mx, s[j][r] * scale);
      mx = redmax16(mx);
      const float mnew = fmaxf(mrun[r], mx);
      const float corr = __expf(mrun[r] - mnew);
      float psum = 0.f;
#pragma unroll
      for (int j = 0; j < 4; ++j) {
        const float e = __expf(s[j][r] * scale - mnew);
        psum += e;
        plds[(r + 8 * hi) * 64 + 16 * j + cn] = f2bf(e);
      }
      psum = redsum16(psum);
      lrun[r] = lrun[r] * corr + psum;
      mrun[r] = mnew;
      o[0][r] *= corr;
      o[1][r] *= corr;
    }
    __syncthreads();
#pragma unroll
    for (int ks = 0; ks < 2; ++ks) {
      const int k0 = ks * 32;
      const v16bf ap = frag_A(plds, 64, k0, lane);
#pragma unroll
      for (int n = 0; n < 2; ++n) {
        v16bf bv = frag_B_KN(kbase + 16 * n, 256, k0, lane);
        o[n] = wmma_bf16(ap, bv, o[n]);
      }
    }
    __syncthreads();
  }
#pragma unroll
  for (int n = 0; n < 2; ++n)
#pragma unroll
    for (int r = 0; r < 8; ++r) {
      const float v = o[n][r] / lrun[r];
      CTX[(size_t)(b * 512 + qt * 16 + r + 8 * hi) * 256 + h * 32 + n * 16 +
          cn] = f2bf(v);
    }
}

// ---------------------------------------------------------------------------
// Final: out = LayerNorm(query + ctx2) ; out rows are (t,b) order
// ---------------------------------------------------------------------------
__global__ __launch_bounds__(256) void final_ln(
    const float* __restrict__ query, const float* __restrict__ ctx2,
    const float* __restrict__ g, const float* __restrict__ bb,
    float* __restrict__ out) {
  __shared__ float red[256];
  const int blk = blockIdx.x, c = threadIdx.x;   // blk = t*2 + b
  const int t = blk >> 1, b = blk & 1;
  const float v = query[(size_t)blk * 256 + c] +
                  ctx2[(size_t)(b * 512 + t) * 256 + c];
  red[c] = v;
  __syncthreads();
#pragma unroll
  for (int s = 128; s > 0; s >>= 1) {
    if (c < s) red[c] += red[c + s];
    __syncthreads();
  }
  const float mean = red[0] * (1.f / 256.f);
  __syncthreads();
  const float d = v - mean;
  red[c] = d * d;
  __syncthreads();
#pragma unroll
  for (int s = 128; s > 0; s >>= 1) {
    if (c < s) red[c] += red[c + s];
    __syncthreads();
  }
  const float var = red[0] * (1.f / 256.f);
  out[(size_t)blk * 256 + c] = d * rsqrtf(var + 1e-5f) * g[c] + bb[c];
}

// ---------------------------------------------------------------------------
// Utility kernels
// ---------------------------------------------------------------------------
__global__ void cast_f32_bf16(const float* __restrict__ s,
                              unsigned short* __restrict__ d, int n) {
  for (int i = blockIdx.x * blockDim.x + threadIdx.x; i < n;
       i += gridDim.x * blockDim.x)
    d[i] = f2bf(s[i]);
}

__global__ void add_plain(float* __restrict__ x, const float* __restrict__ o,
                          int n) {
  for (int i = blockIdx.x * blockDim.x + threadIdx.x; i < n;
       i += gridDim.x * blockDim.x)
    x[i] += o[i];
}

__global__ void add_window_reverse(float* __restrict__ x,
                                   const float* __restrict__ o) {
  const int n = 32768 * 256;
  for (int i = blockIdx.x * blockDim.x + threadIdx.x; i < n;
       i += gridDim.x * blockDim.x) {
    const int row = i >> 8, c = i & 255;
    const int b = row >> 14, l = row & 16383;
    const int yy = l >> 7, xx = l & 127;
    const int wrow = ((b << 8) + ((yy >> 3) << 4) + (xx >> 3)) * 64 +
                     (yy & 7) * 8 + (xx & 7);
    x[i] += o[(size_t)wrow * 256 + c];
  }
}

// query (Tq,B,C) rows t*2+b  ->  bf16 rows b*512+t
__global__ void reorder_query_bf16(const float* __restrict__ q,
                                   unsigned short* __restrict__ d) {
  const int n = 1024 * 256;
  for (int i = blockIdx.x * blockDim.x + threadIdx.x; i < n;
       i += gridDim.x * blockDim.x) {
    const int row = i >> 8, c = i & 255;
    const int t = row >> 1, b = row & 1;
    d[(size_t)((b << 9) + t) * 256 + c] = f2bf(q[i]);
  }
}

// ---------------------------------------------------------------------------
// Host launcher
// ---------------------------------------------------------------------------
extern "C" void kernel_launch(void* const* d_in, const int* in_sizes, int n_in,
                              void* d_out, int out_size, void* d_ws,
                              size_t ws_size, hipStream_t stream) {
  (void)in_sizes; (void)n_in; (void)out_size; (void)ws_size;
  const float* x_in  = (const float*)d_in[0];
  const float* query = (const float*)d_in[1];
  const float* n1g   = (const float*)d_in[2];
  const float* n1b   = (const float*)d_in[3];
  const float* qkv_w = (const float*)d_in[4];
  const float* qkv_b = (const float*)d_in[5];
  const float* out_w = (const float*)d_in[6];
  const float* out_b = (const float*)d_in[7];
  const float* rel_b = (const float*)d_in[8];
  const float* n2g   = (const float*)d_in[9];
  const float* n2b   = (const float*)d_in[10];
  const float* fc1_w = (const float*)d_in[11];
  const float* fc1_b = (const float*)d_in[12];
  const float* fc2_w = (const float*)d_in[13];
  const float* fc2_b = (const float*)d_in[14];
  const float* cng   = (const float*)d_in[15];
  const float* cnb   = (const float*)d_in[16];
  const float* pq_w  = (const float*)d_in[17];
  const float* cf_w  = (const float*)d_in[18];
  const float* cf_b  = (const float*)d_in[19];
  const float* cf1_w = (const float*)d_in[20];
  const float* cf1_b = (const float*)d_in[21];
  const float* cf2_w = (const float*)d_in[22];
  const float* cf2_b = (const float*)d_in[23];
  float* out = (float*)d_out;

  char* ws = (char*)d_ws;
  size_t off = 0;
  auto carve = [&](size_t bytes) -> char* {
    char* p = ws + off;
    off += (bytes + 255) & ~(size_t)255;
    return p;
  };
  float*          XBUF  = (float*)carve((size_t)32768 * 256 * 4);
  unsigned short* AWIN  = (unsigned short*)carve((size_t)32768 * 256 * 2);
  unsigned short* QKVB  = (unsigned short*)carve((size_t)32768 * 768 * 2);
  unsigned short* OATT  = (unsigned short*)carve((size_t)32768 * 256 * 2);
  float*          FOUT  = (float*)carve((size_t)32768 * 256 * 4);
  unsigned short* FC1B  = (unsigned short*)carve((size_t)32768 * 1024 * 2);
  unsigned short* W_QKV = (unsigned short*)carve((size_t)2 * 768 * 256 * 2);
  unsigned short* W_OUT = (unsigned short*)carve((size_t)2 * 256 * 256 * 2);
  unsigned short* W_FC1 = (unsigned short*)carve((size_t)2 * 1024 * 256 * 2);
  unsigned short* W_FC2 = (unsigned short*)carve((size_t)2 * 256 * 1024 * 2);
  unsigned short* W_PQ  = (unsigned short*)carve((size_t)256 * 256 * 2);
  unsigned short* W_CF  = (unsigned short*)carve((size_t)256 * 256 * 2);
  unsigned short* W_CF1 = (unsigned short*)carve((size_t)512 * 256 * 2);
  unsigned short* W_CF2 = (unsigned short*)carve((size_t)256 * 512 * 2);
  unsigned short* AQ    = (unsigned short*)carve((size_t)1024 * 256 * 2);
  unsigned short* QLIN  = (unsigned short*)carve((size_t)1024 * 256 * 2);
  unsigned short* CTXB  = (unsigned short*)carve((size_t)1024 * 256 * 2);
  unsigned short* CB1   = (unsigned short*)carve((size_t)1024 * 256 * 2);
  unsigned short* CB2   = (unsigned short*)carve((size_t)1024 * 512 * 2);
  float*          CTX2F = (float*)carve((size_t)1024 * 256 * 4);
  unsigned short* KVB   = QKVB;  // reuse: QKV buffer free after Swin blocks

  auto cast = [&](const float* s, unsigned short* d, int n) {
    cast_f32_bf16<<<dim3(256), dim3(256), 0, stream>>>(s, d, n);
  };
  cast(qkv_w, W_QKV, 2 * 768 * 256);
  cast(out_w, W_OUT, 2 * 256 * 256);
  cast(fc1_w, W_FC1, 2 * 1024 * 256);
  cast(fc2_w, W_FC2, 2 * 256 * 1024);
  cast(pq_w,  W_PQ,  256 * 256);
  cast(cf_w,  W_CF,  256 * 256);
  cast(cf1_w, W_CF1, 512 * 256);
  cast(cf2_w, W_CF2, 256 * 512);

  hipMemcpyAsync(XBUF, x_in, (size_t)32768 * 256 * 4,
                 hipMemcpyDeviceToDevice, stream);

  for (int d = 0; d < 2; ++d) {
    ln_to_bf16<<<dim3(32768), dim3(256), 0, stream>>>(
        XBUF, n1g + d * 256, n1b + d * 256, AWIN, 1);
    gemm_bf16<0, 0, 1><<<dim3(256, 6), dim3(128), 0, stream>>>(
        AWIN, W_QKV + (size_t)d * 768 * 256, qkv_b + d * 768, nullptr, QKVB,
        32768, 768, 256);
    win_attn<<<dim3(4096), dim3(32), 0, stream>>>(
        QKVB, rel_b + (size_t)d * 225 * 8, OATT);
    gemm_bf16<0, 1, 0><<<dim3(256, 2), dim3(128), 0, stream>>>(
        OATT, W_OUT + (size_t)d * 256 * 256, out_b + d * 256, FOUT, nullptr,
        32768, 256, 256);
    add_window_reverse<<<dim3(1024), dim3(256), 0, stream>>>(XBUF, FOUT);
    ln_to_bf16<<<dim3(32768), dim3(256), 0, stream>>>(
        XBUF, n2g + d * 256, n2b + d * 256, AWIN, 0);
    gemm_bf16<1, 0, 1><<<dim3(256, 8), dim3(128), 0, stream>>>(
        AWIN, W_FC1 + (size_t)d * 1024 * 256, fc1_b + d * 1024, nullptr, FC1B,
        32768, 1024, 256);
    gemm_bf16<0, 1, 0><<<dim3(256, 2), dim3(128), 0, stream>>>(
        FC1B, W_FC2 + (size_t)d * 256 * 1024, fc2_b + d * 256, FOUT, nullptr,
        32768, 256, 1024);
    add_plain<<<dim3(1024), dim3(256), 0, stream>>>(XBUF, FOUT, 32768 * 256);
  }

  // ---- cross attention ----
  cast(XBUF, KVB, 32768 * 256);  // keys/values = final x, cast to bf16
  reorder_query_bf16<<<dim3(1024), dim3(256), 0, stream>>>(query, AQ);
  gemm_bf16<0, 0, 1><<<dim3(8, 2), dim3(128), 0, stream>>>(
      AQ, W_PQ, nullptr, nullptr, QLIN, 1024, 256, 256);
  cross_attn<<<dim3(512), dim3(32), 0, stream>>>(QLIN, KVB, CTXB);
  gemm_bf16<3, 0, 1><<<dim3(8, 2), dim3(128), 0, stream>>>(
      CTXB, W_CF, cf_b, nullptr, CB1, 1024, 256, 256);
  gemm_bf16<2, 0, 1><<<dim3(8, 4), dim3(128), 0, stream>>>(
      CB1, W_CF1, cf1_b, nullptr, CB2, 1024, 512, 256);
  gemm_bf16<0, 1, 0><<<dim3(8, 2), dim3(128), 0, stream>>>(
      CB2, W_CF2, cf2_b, CTX2F, nullptr, 1024, 256, 512);
  final_ln<<<dim3(1024), dim3(256), 0, stream>>>(query, CTX2F, cng, cnb, out);
}
